// PatternMatching_8907762172201
// MI455X (gfx1250) — matrix operand order repaired and checked by reading, not compiled, not run on hardware
//
#include <hip/hip_runtime.h>

typedef __attribute__((ext_vector_type(16))) _Float16 v16h;
typedef __attribute__((ext_vector_type(8)))  float    v8f;

#define NEGV (-1e9f)

// Problem constants (match setup_inputs)
constexpr int B_  = 32;
constexpr int L_  = 512;
constexpr int S1_ = 513;          // number of states (L+1)
constexpr int LP_ = 16;           // pattern length
constexpr int SP_ = 544;          // states padded to multiple of 32
constexpr int KT_ = SP_ / 32;     // 17 K-tiles (WMMA K=32)
constexpr int NT_ = SP_ / 16;     // 34 N-tiles (WMMA N=16)

// Workspace layout (bytes); total ~5.9 MB
constexpr size_t OFF_CS = 0;                                   // cs  [B][3][SP] f32
constexpr size_t SZ_CS  = (size_t)B_ * 3 * SP_ * 4;
constexpr size_t SZ_E   = (size_t)3 * KT_ * NT_ * 32 * 16 * 2; // fragment-packed f16
constexpr size_t OFF_E  = OFF_CS + SZ_CS;                      // E  (forward)
constexpr size_t OFF_ET = OFF_E + SZ_E;                        // E^T (backward)
constexpr size_t SZ_A   = (size_t)B_ * (LP_ + 1) * SP_ * 4;
constexpr size_t OFF_A  = OFF_ET + SZ_E;                       // log-alpha store
constexpr size_t OFF_Bt = OFF_A + SZ_A;                        // log-beta store

// -------------------------------------------------------------------------
// cs[b][c][j] = sum_{i<j} log ss3[b][c][i]  (prefix sums over 96 rows)
// -------------------------------------------------------------------------
__global__ void cs_kernel(const float* __restrict__ ss3, float* __restrict__ cs) {
    int row = blockIdx.x;                 // 0..95 : b*3 + c
    if (threadIdx.x != 0) return;
    const float* src = ss3 + (size_t)row * L_;
    float* dst = cs + (size_t)row * SP_;
    float acc = 0.f;
    dst[0] = 0.f;
    for (int i = 0; i < L_; ++i) { acc += logf(src[i]); dst[i + 1] = acc; }
    for (int j = S1_; j < SP_; ++j) dst[j] = 0.f;  // padding (never read meaningfully)
}

// -------------------------------------------------------------------------
// Build E (and E^T) = exp(Q) with strict upper-triangular mask, stored in
// a contiguous-K B-fragment layout, same K order as uA (32x16: lanes 0-15 hold K=0..15,
// lanes 16-31 hold K=16..31; 16 ascending-K halves per lane).
// -------------------------------------------------------------------------
__global__ void pack_kernel(const float* __restrict__ Q,
                            _Float16* __restrict__ E, _Float16* __restrict__ ET) {
    long idx = (long)blockIdx.x * blockDim.x + threadIdx.x;
    long total = (long)3 * SP_ * SP_;
    if (idx >= total) return;
    int c  = (int)(idx / ((long)SP_ * SP_));
    int r  = (int)(idx % ((long)SP_ * SP_));
    int kk = r / SP_;   // K-dim index (contraction)
    int nn = r % SP_;   // N-dim index (output)

    int kt = kk >> 5, kl = kk & 31;
    int nt = nn >> 4, nl = nn & 15;
    int lane = nl + ((kl & 16) ? 16 : 0);
    int hidx = kl & 15;
    size_t addr = ((((size_t)c * KT_ + kt) * NT_ + nt) * 32 + lane) * 16 + hidx;

    // forward:  K = j (=kk), N = k (=nn); nonzero iff j < k, both < S1
    float ev = 0.f;
    if (kk < nn && nn < S1_)
        ev = expf(Q[((size_t)c * S1_ + kk) * S1_ + nn]);
    E[addr] = (_Float16)ev;

    // backward: K = k (=kk), N = j (=nn); nonzero iff j < k, both < S1
    float etv = 0.f;
    if (nn < kk && kk < S1_)
        etv = expf(Q[((size_t)c * S1_ + nn) * S1_ + kk]);
    ET[addr] = (_Float16)etv;
}

// -------------------------------------------------------------------------
// Scan kernel: 4 blocks = {fwd,bwd} x {batch tile 0,1}, 256 threads (8 waves).
// Each step: row-max -> exp into A-fragment LDS -> WMMA GEMM chain -> log.
// -------------------------------------------------------------------------
__global__ __launch_bounds__(256) void scan_kernel(
        const float* __restrict__ cs,
        const _Float16* __restrict__ E, const _Float16* __restrict__ ET,
        const int* __restrict__ pattern, const int* __restrict__ ls,
        float* __restrict__ aS, float* __restrict__ bS) {

    __shared__ float    alpha[16 * SP_];        // current log-state, 34.8 KB
    __shared__ _Float16 uA[KT_ * 32 * 16];      // A fragments (exp domain), 17.4 KB
    __shared__ float    partial[16 * 16];
    __shared__ float    rowmax[16];

    const int tid  = threadIdx.x;
    const int dir  = blockIdx.x >> 1;           // 0 = forward, 1 = backward
    const int bt   = blockIdx.x & 1;            // batch tile (16 batches)
    const int wave = tid >> 5, lane = tid & 31;
    const float s1 = dir ? 1.f : -1.f;          // phase1: g = state + s1*cs
    float* out = dir ? bS : aS;

    // ---- init state + store boundary slice ----
    for (int i = tid; i < 16 * SP_; i += 256) {
        int m = i / SP_, j = i % SP_;
        int gb = bt * 16 + m;
        float v;
        if (dir == 0) v = (j == 0)      ? 0.f : NEGV;
        else          v = (j == ls[gb]) ? 0.f : NEGV;
        alpha[i] = v;
        int step0 = dir ? LP_ : 0;
        out[((size_t)gb * (LP_ + 1) + step0) * SP_ + j] = v;
    }
    __syncthreads();

    for (int t = 0; t < LP_; ++t) {
        const int c = pattern[dir ? (LP_ - 1 - t) : t];
        const int m = tid >> 4, sub = tid & 15;
        const int gbRow = bt * 16 + m;
        const float* csRow = cs + ((size_t)gbRow * 3 + c) * SP_;

        // phase 1a: per-batch-row max of g = state + s1*cs
        float pm = NEGV;
        for (int j = sub; j < S1_; j += 16)
            pm = fmaxf(pm, alpha[m * SP_ + j] + s1 * csRow[j]);
        partial[m * 16 + sub] = pm;
        __syncthreads();
        if (sub == 0) {
            float mx = partial[m * 16];
            for (int q = 1; q < 16; ++q) mx = fmaxf(mx, partial[m * 16 + q]);
            rowmax[m] = mx;
        }
        __syncthreads();

        // phase 1b: u = exp(g - rowmax), scattered into A-fragment layout
        const float mrow = rowmax[m];
        for (int j = sub; j < SP_; j += 16) {
            _Float16 h = (_Float16)0.f;
            if (j < S1_) {
                float g = alpha[m * SP_ + j] + s1 * csRow[j] - mrow;
                h = (_Float16)expf(g);
            }
            int kt = j >> 5, jl = j & 31;
            int fl = m + ((jl & 16) ? 16 : 0);            // fragment lane
            int fi = jl & 15;                             // half index in lane
            uA[(kt * 32 + fl) * 16 + fi] = h;
        }
        __syncthreads();

        // phase 2+3: v = u @ E (WMMA chain over K), then log back to log-space
        const _Float16* Ec = (dir ? ET : E) + (size_t)c * KT_ * NT_ * 32 * 16;
        for (int nt = wave; nt < NT_; nt += 8) {
            v8f acc = {};
            for (int kt = 0; kt < KT_; ++kt) {
                v16h af = *(const v16h*)&uA[(kt * 32 + lane) * 16];
                v16h bf = *(const v16h*)&Ec[((size_t)(kt * NT_ + nt) * 32 + lane) * 16];
                acc = __builtin_amdgcn_wmma_f32_16x16x32_f16(
                        false, af, false, bf, (short)0, acc, false, false);
            }
            const int n   = nt * 16 + (lane & 15);
            const int mhi = (lane >> 4) * 8;
            for (int r = 0; r < 8; ++r) {
                int mloc = mhi + r;
                int gb = bt * 16 + mloc;
                float v = acc[r];
                float anew = NEGV;
                if (n < S1_ && v > 0.f) {
                    float csv = cs[((size_t)gb * 3 + c) * SP_ + n];
                    anew = -s1 * csv + rowmax[mloc] + logf(v);
                }
                alpha[mloc * SP_ + n] = anew;
                int stepIdx = dir ? (LP_ - 1 - t) : (t + 1);
                out[((size_t)gb * (LP_ + 1) + stepIdx) * SP_ + n] = anew;
            }
        }
        __syncthreads();
    }
}

// -------------------------------------------------------------------------
// p[b,t,s] = a[b,t,s] + beta[b,t,s] - a[b,LP,ls[b]]
// -------------------------------------------------------------------------
__global__ void combine_kernel(const float* __restrict__ aS, const float* __restrict__ bS,
                               const int* __restrict__ ls, float* __restrict__ outp, int total) {
    int i = blockIdx.x * blockDim.x + threadIdx.x;
    if (i >= total) return;
    int b = i / ((LP_ + 1) * S1_);
    int r = i % ((LP_ + 1) * S1_);
    int step = r / S1_, s = r % S1_;
    float M = aS[((size_t)b * (LP_ + 1) + LP_) * SP_ + ls[b]];
    outp[i] = aS[((size_t)b * (LP_ + 1) + step) * SP_ + s]
            + bS[((size_t)b * (LP_ + 1) + step) * SP_ + s] - M;
}

extern "C" void kernel_launch(void* const* d_in, const int* in_sizes, int n_in,
                              void* d_out, int out_size, void* d_ws, size_t ws_size,
                              hipStream_t stream) {
    (void)in_sizes; (void)n_in; (void)ws_size;
    const float* ss3     = (const float*)d_in[0];   // [B,3,L]
    const float* Q       = (const float*)d_in[1];   // [3,S1,S1]
    const int*   pattern = (const int*)  d_in[2];   // [LP]
    const int*   ls      = (const int*)  d_in[3];   // [B]

    char* ws = (char*)d_ws;                         // needs ~6.2 MB
    float*    cs = (float*)   (ws + OFF_CS);
    _Float16* E  = (_Float16*)(ws + OFF_E);
    _Float16* ET = (_Float16*)(ws + OFF_ET);
    float*    aS = (float*)   (ws + OFF_A);
    float*    bS = (float*)   (ws + OFF_Bt);

    cs_kernel<<<96, 32, 0, stream>>>(ss3, cs);

    long totalPack = (long)3 * SP_ * SP_;
    pack_kernel<<<(int)((totalPack + 255) / 256), 256, 0, stream>>>(Q, E, ET);

    scan_kernel<<<4, 256, 0, stream>>>(cs, E, ET, pattern, ls, aS, bS);

    int totalOut = B_ * (LP_ + 1) * S1_;
    combine_kernel<<<(totalOut + 255) / 256, 256, 0, stream>>>(aS, bS, ls, (float*)d_out, totalOut);
}
